// MultiHeadAttention_19516331393542
// MI455X (gfx1250) — compile-verified, matrix operand order checked by reading, not run on hardware
//
#include <hip/hip_runtime.h>

typedef _Float16 h16;
typedef __attribute__((ext_vector_type(8)))  _Float16 v8h_t;
typedef __attribute__((ext_vector_type(16))) _Float16 v16h;
typedef __attribute__((ext_vector_type(8)))  float v8f;
typedef __attribute__((ext_vector_type(4)))  float v4f;

#define HID 1024
#define SEQ 2048
#define NB  2
#define NH  16
#define DH  64
#define TOK (NB*SEQ)   /* 4096 tokens */
#define LP  2056       /* padded LDS row stride (halves): 16B-aligned, 4-bank row skew */

// D = A(16x32 f16) * B(32x16 f16) + C(16x16 f32)
__device__ __forceinline__ v8f wmma16(v16h a, v16h b, v8f c) {
  return __builtin_amdgcn_wmma_f32_16x16x32_f16(false, a, false, b, (short)0, c, false, false);
}

// Fragment load from f16 memory per CDNA5 layout: p[0..7] -> [0..7], p[16..23] -> [8..15]
__device__ __forceinline__ v16h frag_h16(const h16* p) {
  v8h_t lo = *(const v8h_t*)(p);
  v8h_t hi = *(const v8h_t*)(p + 16);
  v16h a;
#pragma unroll
  for (int i = 0; i < 8; i++) { a[i] = lo[i]; a[8 + i] = hi[i]; }
  return a;
}

// ---------------- Kernel 0: one-time f32 -> f16 pack (query + weights) -------------
__global__ __launch_bounds__(256)
void pack_f16_kernel(const float* __restrict__ src, h16* __restrict__ dst, int n8)
{
  const int i = blockIdx.x * 256 + threadIdx.x;
  if (i < n8) {
    const v4f a = *(const v4f*)(src + (size_t)i * 8);
    const v4f b = *(const v4f*)(src + (size_t)i * 8 + 4);
    v8h_t o;
#pragma unroll
    for (int e = 0; e < 4; e++) { o[e] = (_Float16)a[e]; o[4 + e] = (_Float16)b[e]; }
    *(v8h_t*)(dst + (size_t)i * 8) = o;
  }
}

// ---------------- Kernel 1: fused Q/K/V projections (z selects matrix) -------------
// Wave computes 32(M tokens) x 64(N features); K=1024 in 32-steps; all-f16 operands.
__global__ __launch_bounds__(128)
void qkv_proj_kernel(const h16* __restrict__ qf,
                     const h16* __restrict__ Wq16, const h16* __restrict__ Wk16,
                     const h16* __restrict__ Wv16,
                     const float* __restrict__ bq, const float* __restrict__ bk,
                     const float* __restrict__ bv,
                     h16* __restrict__ qbuf, h16* __restrict__ kbuf, h16* __restrict__ vtb)
{
  const int z = blockIdx.z;
  const h16*   W  = (z == 0) ? Wq16 : ((z == 1) ? Wk16 : Wv16);
  const float* bs = (z == 0) ? bq : ((z == 1) ? bk : bv);

  const int lane = threadIdx.x & 31, wave = threadIdx.x >> 5;
  const int lidx = lane & 15, lhalf = lane >> 4;
  const int m0 = blockIdx.y * 128 + wave * 32;
  const int n0 = blockIdx.x * 64;

  const h16* ar0 = qf + (size_t)(m0 + lidx) * HID;
  const h16* ar1 = qf + (size_t)(m0 + 16 + lidx) * HID;
  const h16* br0 = W + (size_t)(n0 +  0 + lidx) * HID;
  const h16* br1 = W + (size_t)(n0 + 16 + lidx) * HID;
  const h16* br2 = W + (size_t)(n0 + 32 + lidx) * HID;
  const h16* br3 = W + (size_t)(n0 + 48 + lidx) * HID;

  v8f acc[2][4] = {};
  for (int k0 = 0; k0 < HID; k0 += 32) {
    const int ka = k0 + lhalf * 8;
    const v16h a0 = frag_h16(ar0 + ka);
    const v16h a1 = frag_h16(ar1 + ka);
    const v16h b0 = frag_h16(br0 + ka);
    const v16h b1 = frag_h16(br1 + ka);
    const v16h b2 = frag_h16(br2 + ka);
    const v16h b3 = frag_h16(br3 + ka);
    acc[0][0] = wmma16(a0, b0, acc[0][0]);
    acc[0][1] = wmma16(a0, b1, acc[0][1]);
    acc[0][2] = wmma16(a0, b2, acc[0][2]);
    acc[0][3] = wmma16(a0, b3, acc[0][3]);
    acc[1][0] = wmma16(a1, b0, acc[1][0]);
    acc[1][1] = wmma16(a1, b1, acc[1][1]);
    acc[1][2] = wmma16(a1, b2, acc[1][2]);
    acc[1][3] = wmma16(a1, b3, acc[1][3]);
  }

  const int rbase = lhalf * 8;
#pragma unroll
  for (int ms = 0; ms < 2; ms++) {
    const int mrow = m0 + ms * 16;
#pragma unroll
    for (int f = 0; f < 4; f++) {
      const int n = n0 + f * 16 + lidx;
      const float bval = bs[n];
      if (z == 2) {
        // V stored transposed: vt[b][feature n][token l], 8 contiguous tokens/lane
        const int bb = mrow >> 11;
        const int l0 = (mrow & 2047) + rbase;
        v8h_t pk;
#pragma unroll
        for (int r = 0; r < 8; r++) pk[r] = (_Float16)(acc[ms][f][r] + bval);
        *(v8h_t*)(vtb + ((size_t)(bb * HID + n)) * SEQ + l0) = pk;
      } else {
        h16* dst = (z == 0) ? qbuf : kbuf;
        const float sc = (z == 0) ? 0.125f : 1.0f; // q pre-scaled by 1/sqrt(64)
#pragma unroll
        for (int r = 0; r < 8; r++)
          dst[(size_t)(mrow + rbase + r) * HID + n] = (_Float16)((acc[ms][f][r] + bval) * sc);
      }
    }
  }
}

// ---------------- Kernel 2: attention for one (b, h, 16 q-rows) per workgroup ------
__global__ __launch_bounds__(128)
void attn_kernel(const h16* __restrict__ qb, const h16* __restrict__ kb,
                 const h16* __restrict__ vt, const float* __restrict__ bias,
                 h16* __restrict__ xb)
{
  __shared__ h16   S[16 * LP];            // 64.25 KB: logits then probabilities (f16)
  __shared__ float Opart[4][16 * 64];     // 16 KB: per-wave PV partials
  __shared__ float red[16][8];
  __shared__ float rowmax[16];
  __shared__ float rowsum[16];

  const int tid = threadIdx.x, lane = tid & 31, wave = tid >> 5;
  const int lidx = lane & 15, lhalf = lane >> 4;
  const int bh = blockIdx.y, b = bh >> 4, h = bh & 15;
  const int q0 = blockIdx.x * 16;
  const size_t tokq = (size_t)(b * SEQ + q0);

  // Q fragments (K-dim = 64 head dims -> two 16x32 A-frags), resident in VGPRs
  const h16* qrow = qb + (tokq + lidx) * HID + h * DH;
  const v16h aq0 = frag_h16(qrow + lhalf * 8);
  const v16h aq1 = frag_h16(qrow + 32 + lhalf * 8);

  // ---- S = Q K^T + bias: wave w owns keys [w*512, w*512+512) ----
  const float* brow = bias + b * SEQ;
  for (int j = 0; j < 32; j++) {
    const int n0 = wave * 512 + j * 16;
    const h16* krow = kb + ((size_t)(b * SEQ + n0 + lidx)) * HID + h * DH;
    v8f c = {};
    c = wmma16(aq0, frag_h16(krow + lhalf * 8), c);
    c = wmma16(aq1, frag_h16(krow + 32 + lhalf * 8), c);
    const float bval = brow[n0 + lidx];
#pragma unroll
    for (int r = 0; r < 8; r++)
      S[(size_t)(r + 8 * lhalf) * LP + n0 + lidx] = (_Float16)(c[r] + bval);
  }
  __syncthreads();

  // ---- exact two-pass softmax: thread -> (row = tid&15, 256-key segment = tid>>4) ----
  const int row = tid & 15, seg = tid >> 4;
  h16* srow = S + (size_t)row * LP + seg * 256;
  float mx = -3.0e38f;
  for (int i = 0; i < 256; i += 8) {
    v8h_t v = *(const v8h_t*)(srow + i);
#pragma unroll
    for (int e = 0; e < 8; e++) mx = fmaxf(mx, (float)v[e]);
  }
  red[row][seg] = mx;
  __syncthreads();
  if (tid < 16) {
    float m = red[tid][0];
#pragma unroll
    for (int s = 1; s < 8; s++) m = fmaxf(m, red[tid][s]);
    rowmax[tid] = m;
  }
  __syncthreads();
  const float m = rowmax[row];
  float sum = 0.f;
  for (int i = 0; i < 256; i += 8) {
    v8h_t v = *(const v8h_t*)(srow + i);
    v8h_t o;
#pragma unroll
    for (int e = 0; e < 8; e++) { float ev = __expf((float)v[e] - m); sum += ev; o[e] = (_Float16)ev; }
    *(v8h_t*)(srow + i) = o;   // P overwrites S in place (f16)
  }
  red[row][seg] = sum;
  __syncthreads();
  if (tid < 16) {
    float s = 0.f;
#pragma unroll
    for (int s8 = 0; s8 < 8; s8++) s += red[tid][s8];
    rowsum[tid] = s;
  }
  __syncthreads();

  // ---- O_partial = P * V over this wave's key range (V^T rows contiguous in keys) ----
  v8f o0 = {}, o1 = {}, o2 = {}, o3 = {};
  const h16* v0 = vt + ((size_t)(b * HID + h * DH +  0 + lidx)) * SEQ;
  const h16* v1 = vt + ((size_t)(b * HID + h * DH + 16 + lidx)) * SEQ;
  const h16* v2 = vt + ((size_t)(b * HID + h * DH + 32 + lidx)) * SEQ;
  const h16* v3 = vt + ((size_t)(b * HID + h * DH + 48 + lidx)) * SEQ;
  for (int kk = wave * 512; kk < wave * 512 + 512; kk += 32) {
    const int kbase = kk + lhalf * 8;
    v16h ap = frag_h16(S + (size_t)lidx * LP + kbase);
    o0 = wmma16(ap, frag_h16(v0 + kbase), o0);
    o1 = wmma16(ap, frag_h16(v1 + kbase), o1);
    o2 = wmma16(ap, frag_h16(v2 + kbase), o2);
    o3 = wmma16(ap, frag_h16(v3 + kbase), o3);
  }
  {
    v8f oo[4] = {o0, o1, o2, o3};
#pragma unroll
    for (int f = 0; f < 4; f++)
#pragma unroll
      for (int r = 0; r < 8; r++)
        Opart[wave][(r + 8 * lhalf) * 64 + f * 16 + lidx] = oo[f][r];
  }
  __syncthreads();

  // ---- cross-wave reduce + 1/rowsum scale, f16 store to x buffer [B,L,1024] ----
  {
    const int e0 = tid * 8;
    const int rrow = e0 >> 6, col = e0 & 63;
    const float inv = 1.0f / rowsum[rrow];
    v8h_t ov;
#pragma unroll
    for (int e = 0; e < 8; e++) {
      float s = Opart[0][e0 + e] + Opart[1][e0 + e] + Opart[2][e0 + e] + Opart[3][e0 + e];
      ov[e] = (_Float16)(s * inv);
    }
    *(v8h_t*)(xb + (tokq + rrow) * HID + h * DH + col) = ov;
  }
}

// ---------------- Kernel 3: output projection, f32 result ------------------------
__global__ __launch_bounds__(128)
void out_proj_kernel(const h16* __restrict__ xb, const h16* __restrict__ Wo16,
                     const float* __restrict__ bo, float* __restrict__ out)
{
  const int lane = threadIdx.x & 31, wave = threadIdx.x >> 5;
  const int lidx = lane & 15, lhalf = lane >> 4;
  const int m0 = blockIdx.y * 128 + wave * 32;
  const int n0 = blockIdx.x * 64;

  const h16* ar0 = xb + (size_t)(m0 + lidx) * HID;
  const h16* ar1 = xb + (size_t)(m0 + 16 + lidx) * HID;
  const h16* br0 = Wo16 + (size_t)(n0 +  0 + lidx) * HID;
  const h16* br1 = Wo16 + (size_t)(n0 + 16 + lidx) * HID;
  const h16* br2 = Wo16 + (size_t)(n0 + 32 + lidx) * HID;
  const h16* br3 = Wo16 + (size_t)(n0 + 48 + lidx) * HID;

  v8f acc[2][4] = {};
  for (int k0 = 0; k0 < HID; k0 += 32) {
    const int ka = k0 + lhalf * 8;
    const v16h a0 = frag_h16(ar0 + ka);
    const v16h a1 = frag_h16(ar1 + ka);
    const v16h b0 = frag_h16(br0 + ka);
    const v16h b1 = frag_h16(br1 + ka);
    const v16h b2 = frag_h16(br2 + ka);
    const v16h b3 = frag_h16(br3 + ka);
    acc[0][0] = wmma16(a0, b0, acc[0][0]);
    acc[0][1] = wmma16(a0, b1, acc[0][1]);
    acc[0][2] = wmma16(a0, b2, acc[0][2]);
    acc[0][3] = wmma16(a0, b3, acc[0][3]);
    acc[1][0] = wmma16(a1, b0, acc[1][0]);
    acc[1][1] = wmma16(a1, b1, acc[1][1]);
    acc[1][2] = wmma16(a1, b2, acc[1][2]);
    acc[1][3] = wmma16(a1, b3, acc[1][3]);
  }

  const int rbase = lhalf * 8;
#pragma unroll
  for (int ms = 0; ms < 2; ms++) {
    const int mrow = m0 + ms * 16;
#pragma unroll
    for (int f = 0; f < 4; f++) {
      const int n = n0 + f * 16 + lidx;
      const float bval = bo[n];
#pragma unroll
      for (int r = 0; r < 8; r++)
        out[(size_t)(mrow + rbase + r) * HID + n] = acc[ms][f][r] + bval;
    }
  }
}

extern "C" void kernel_launch(void* const* d_in, const int* in_sizes, int n_in,
                              void* d_out, int out_size, void* d_ws, size_t ws_size,
                              hipStream_t stream)
{
  (void)in_sizes; (void)n_in; (void)out_size; (void)ws_size;
  const float* query = (const float*)d_in[0];
  const float* bias  = (const float*)d_in[1];
  const float* Wq = (const float*)d_in[2]; const float* bq = (const float*)d_in[3];
  const float* Wk = (const float*)d_in[4]; const float* bk = (const float*)d_in[5];
  const float* Wv = (const float*)d_in[6]; const float* bv = (const float*)d_in[7];
  const float* Wo = (const float*)d_in[8]; const float* bo = (const float*)d_in[9];
  float* out = (float*)d_out;

  char* ws = (char*)d_ws;
  const size_t QSZ = (size_t)TOK * HID * sizeof(h16);   // 8 MB
  const size_t WSZ = (size_t)HID * HID * sizeof(h16);   // 2 MB
  h16* qf16 = (h16*)(ws);
  h16* Wq16 = (h16*)(ws + QSZ);
  h16* Wk16 = (h16*)(ws + QSZ + WSZ);
  h16* Wv16 = (h16*)(ws + QSZ + 2 * WSZ);
  h16* Wo16 = (h16*)(ws + QSZ + 3 * WSZ);
  h16* qbuf = (h16*)(ws + QSZ + 4 * WSZ);
  h16* kbuf = (h16*)(ws + 2 * QSZ + 4 * WSZ);
  h16* vtb  = (h16*)(ws + 3 * QSZ + 4 * WSZ);
  h16* xbuf = (h16*)(ws + 4 * QSZ + 4 * WSZ);

  // one-time f32 -> f16 packs
  const int nq8 = TOK * HID / 8;       // 524288
  const int nw8 = HID * HID / 8;       // 131072
  pack_f16_kernel<<<(nq8 + 255) / 256, 256, 0, stream>>>(query, qf16, nq8);
  pack_f16_kernel<<<(nw8 + 255) / 256, 256, 0, stream>>>(Wq, Wq16, nw8);
  pack_f16_kernel<<<(nw8 + 255) / 256, 256, 0, stream>>>(Wk, Wk16, nw8);
  pack_f16_kernel<<<(nw8 + 255) / 256, 256, 0, stream>>>(Wv, Wv16, nw8);
  pack_f16_kernel<<<(nw8 + 255) / 256, 256, 0, stream>>>(Wo, Wo16, nw8);

  dim3 g1(HID / 64, TOK / 128, 3);
  qkv_proj_kernel<<<g1, 128, 0, stream>>>(qf16, Wq16, Wk16, Wv16, bq, bk, bv,
                                          qbuf, kbuf, vtb);

  dim3 g2(SEQ / 16, NB * NH);
  attn_kernel<<<g2, 128, 0, stream>>>(qbuf, kbuf, vtb, bias, xbuf);

  dim3 g3(HID / 64, TOK / 128);
  out_proj_kernel<<<g3, 128, 0, stream>>>(xbuf, Wo16, bo, out);
}
